// EfficientAttention_13417477833317
// MI455X (gfx1250) — matrix operand order repaired and checked by reading, not compiled, not run on hardware
//
#include <hip/hip_runtime.h>
#include <math.h>

typedef _Float16 f16;
typedef __attribute__((ext_vector_type(16))) _Float16 v16h;
typedef __attribute__((ext_vector_type(8)))  _Float16 v8h;
typedef __attribute__((ext_vector_type(8)))  float    v8f;
typedef __attribute__((ext_vector_type(4)))  unsigned int u32x4;
typedef __attribute__((ext_vector_type(8)))  int i32x8;
typedef __attribute__((ext_vector_type(4)))  int i32x4;

// LDS tile geometry: 128 rows x 64 f16 (128B) + 16B TDM pad -> 144B row stride
#define LDS_ROW_ELEMS 72
#define LDS_TILE_ELEMS (128 * LDS_ROW_ELEMS)   // 9216 f16 = 18432 B per tile

// ---------------------------------------------------------------------------
// TDM: issue a 2D tile load (tile 64 x 128 rows of f16) global -> LDS.
// D# per CDNA5 ISA. Pad 4 dwords every 32 dwords
// (pad_interval=4 -> 32 dwords = one 128B row, pad_amount=3 -> 16B).
// amdgpu-toolchain (clang-23) 6-arg builtin form.
// ---------------------------------------------------------------------------
__device__ __forceinline__ void tdm_load_tile(unsigned lds_off,
                                              const f16* gptr,
                                              unsigned tensor_d0,
                                              unsigned tensor_d1,
                                              unsigned long long stride0) {
  unsigned long long ga = (unsigned long long)(uintptr_t)gptr;
  u32x4 g0;
  g0[0] = 1u;                                            // count=1, user mode
  g0[1] = lds_off;                                       // LDS byte address
  g0[2] = (unsigned)ga;                                  // global_addr[31:0]
  g0[3] = (unsigned)((ga >> 32) & 0x01ffffffu) | (2u << 30);  // addr[56:32]|type=2

  i32x8 g1;
  // data_size=1 (2B), pad_enable, pad_interval=4 (32 dw), pad_amount=3 (4 dw)
  g1[0] = (int)((1u << 16) | (1u << 20) | (4u << 22) | (3u << 25));
  g1[1] = (int)((tensor_d0 & 0xffffu) << 16);            // tensor_dim0 lo16
  g1[2] = (int)((tensor_d0 >> 16) | ((tensor_d1 & 0xffffu) << 16));
  g1[3] = (int)((tensor_d1 >> 16) | (64u << 16));        // tile_dim0 = 64
  g1[4] = (int)(128u);                                   // tile_dim1=128, tile_dim2=0
  g1[5] = (int)(stride0 & 0xffffffffu);                  // dim0 stride lo32
  g1[6] = (int)((stride0 >> 32) & 0xffffu);              // stride hi16, s1 lo16=0
  g1[7] = 0;

  i32x4 z4 = {0, 0, 0, 0};
  i32x8 z8 = {0, 0, 0, 0, 0, 0, 0, 0};
  __builtin_amdgcn_tensor_load_to_lds(g0, g1, z4, z4, z8, 0);
}

// ---------------------------------------------------------------------------
// WMMA fragment loaders from LDS (gfx1250 wave32 layouts, 16x16x32 f16)
// ---------------------------------------------------------------------------
__device__ __forceinline__ v16h frag_a_lds(const f16* base) {
  int l = threadIdx.x & 31;
  int m = l & 15, g = l >> 4;
  const f16* p = base + m * LDS_ROW_ELEMS + g * 8;
  v8h lo = *(const v8h*)p;
  v8h hi = *(const v8h*)(p + 16);
  v16h r;
#pragma unroll
  for (int i = 0; i < 8; ++i) { r[i] = lo[i]; r[i + 8] = hi[i]; }
  return r;
}

__device__ __forceinline__ v16h frag_b_lds(const f16* base) {
  int l = threadIdx.x & 31;
  int n = l & 15, g = l >> 4;
  const f16* p = base + n * LDS_ROW_ELEMS + g * 8;
  v8h lo = *(const v8h*)p;
  v8h hi = *(const v8h*)(p + 16);
  v16h r;
#pragma unroll
  for (int i = 0; i < 8; ++i) { r[i] = lo[i]; r[i + 8] = hi[i]; }
  return r;
}

// ---------------------------------------------------------------------------
// Generic batched TN GEMM:  D[z] = A[z] (MxK, lda) * Bt[z]^T (Bt is NdxK, ldb)
// TDM double-buffered LDS staging; v_wmma_f32_16x16x32_f16 fp32 accumulate.
// Block = 256 threads (8 waves); block tile 128x128; wave tile 32x64.
// Requires M%128==0, Nd%128==0, K%64==0 (true for all call sites here).
// ---------------------------------------------------------------------------
template <bool OUT_F16, bool BIAS>
__global__ __launch_bounds__(256) void gemm_tn(
    const f16* __restrict__ A, int lda, long long sAo, long long sAi,
    const f16* __restrict__ Bt, int ldb, long long sBo, long long sBi,
    void* __restrict__ Dv, int ldd, long long sDo, long long sDi,
    const float* __restrict__ bias, int M, int Nd, int K, int innerN) {
  __shared__ __align__(16) f16 lds[2][2][LDS_TILE_ELEMS];  // [buf][A|B]

  int z = blockIdx.z;
  int zo = z / innerN, zi = z % innerN;
  A  += zo * sAo + zi * sAi;
  Bt += zo * sBo + zi * sBi;
  long long doff = (long long)zo * sDo + (long long)zi * sDi;

  int w  = threadIdx.x >> 5;
  int wm = w & 3, wn = w >> 2;
  int row0 = blockIdx.y * 128 + wm * 32;
  int col0 = blockIdx.x * 128 + wn * 64;

  const f16* Atile = A + (long long)(blockIdx.y * 128) * lda;   // + k0
  const f16* Btile = Bt + (long long)(blockIdx.x * 128) * ldb;  // + k0

  v8f zero = {0.f, 0.f, 0.f, 0.f, 0.f, 0.f, 0.f, 0.f};
  v8f acc[2][4];
#pragma unroll
  for (int i = 0; i < 2; ++i)
#pragma unroll
    for (int j = 0; j < 4; ++j) acc[i][j] = zero;

  bool w0 = (threadIdx.x < 32);
  int nslab = K >> 6;  // K/64

  // Prologue: slab 0 into buffer 0
  if (w0) {
    tdm_load_tile((unsigned)(uintptr_t)&lds[0][0][0], Atile, K, M, lda);
    tdm_load_tile((unsigned)(uintptr_t)&lds[0][1][0], Btile, K, Nd, ldb);
    __builtin_amdgcn_s_wait_tensorcnt(0);
  }
  __syncthreads();

  for (int s = 0; s < nslab; ++s) {
    int cur = s & 1;
    if (w0 && (s + 1) < nslab) {
      int k0 = (s + 1) << 6;
      tdm_load_tile((unsigned)(uintptr_t)&lds[cur ^ 1][0][0], Atile + k0, K, M, lda);
      tdm_load_tile((unsigned)(uintptr_t)&lds[cur ^ 1][1][0], Btile + k0, K, Nd, ldb);
    }
    const f16* Asl = &lds[cur][0][0];
    const f16* Bsl = &lds[cur][1][0];
#pragma unroll
    for (int kk = 0; kk < 64; kk += 32) {
      v16h a0 = frag_a_lds(Asl + (wm * 32) * LDS_ROW_ELEMS + kk);
      v16h a1 = frag_a_lds(Asl + (wm * 32 + 16) * LDS_ROW_ELEMS + kk);
      v16h b[4];
#pragma unroll
      for (int j = 0; j < 4; ++j)
        b[j] = frag_b_lds(Bsl + (wn * 64 + j * 16) * LDS_ROW_ELEMS + kk);
#pragma unroll
      for (int j = 0; j < 4; ++j) {
        acc[0][j] = __builtin_amdgcn_wmma_f32_16x16x32_f16(
            false, a0, false, b[j], (short)0, acc[0][j], false, false);
        acc[1][j] = __builtin_amdgcn_wmma_f32_16x16x32_f16(
            false, a1, false, b[j], (short)0, acc[1][j], false, false);
      }
    }
    if (w0) __builtin_amdgcn_s_wait_tensorcnt(0);
    __syncthreads();
  }

  int l = threadIdx.x & 31;
  int n = l & 15, g = l >> 4;
#pragma unroll
  for (int i = 0; i < 2; ++i) {
#pragma unroll
    for (int j = 0; j < 4; ++j) {
#pragma unroll
      for (int r = 0; r < 8; ++r) {
        int row = row0 + i * 16 + g * 8 + r;
        int col = col0 + j * 16 + n;
        float v = acc[i][j][r];
        if (BIAS) v += bias[col];
        if (OUT_F16)
          ((f16*)Dv)[doff + (long long)row * ldd + col] = (f16)v;
        else
          ((float*)Dv)[doff + (long long)row * ldd + col] = v;
      }
    }
  }
}

// ---------------------------------------------------------------------------
// Elementwise fp32 -> f16 convert
// ---------------------------------------------------------------------------
__global__ void cvt_f32_f16(const float* __restrict__ in, f16* __restrict__ out,
                            long long n) {
  long long i = (long long)blockIdx.x * blockDim.x + threadIdx.x;
  long long stride = (long long)gridDim.x * blockDim.x;
  for (; i < n; i += stride) out[i] = (f16)in[i];
}

// ---------------------------------------------------------------------------
// Row softmax over contiguous 128 values (q path). One wave per row.
// ---------------------------------------------------------------------------
__global__ __launch_bounds__(256) void row_softmax_128(
    const float* __restrict__ in, f16* __restrict__ out, long long nrows) {
  long long row = (long long)blockIdx.x * (blockDim.x >> 5) + (threadIdx.x >> 5);
  if (row >= nrows) return;
  int lane = threadIdx.x & 31;
  const float* p = in + row * 128;
  float v[4], mx = -INFINITY;
#pragma unroll
  for (int i = 0; i < 4; ++i) { v[i] = p[lane + 32 * i]; mx = fmaxf(mx, v[i]); }
#pragma unroll
  for (int off = 16; off; off >>= 1) mx = fmaxf(mx, __shfl_xor(mx, off, 32));
  float s = 0.f;
#pragma unroll
  for (int i = 0; i < 4; ++i) { v[i] = __expf(v[i] - mx); s += v[i]; }
#pragma unroll
  for (int off = 16; off; off >>= 1) s += __shfl_xor(s, off, 32);
  float r = 1.f / s;
  f16* q = out + row * 128;
#pragma unroll
  for (int i = 0; i < 4; ++i) q[lane + 32 * i] = (f16)(v[i] * r);
}

// ---------------------------------------------------------------------------
// Column stats for k softmax over tokens: per (b, channel) max & 1/sum(exp).
// k laid out [4][8192][1024]; 4096 columns total, one thread each.
// ---------------------------------------------------------------------------
__global__ void col_stats(const float* __restrict__ k, float* __restrict__ cmax,
                          float* __restrict__ crsum) {
  int cg = blockIdx.x * blockDim.x + threadIdx.x;  // 0..4095
  const float* p = k + (long long)(cg >> 10) * 8192 * 1024 + (cg & 1023);
  float mx = -INFINITY;
  for (int n0 = 0; n0 < 8192; ++n0) mx = fmaxf(mx, p[(long long)n0 * 1024]);
  float s = 0.f;
  for (int n0 = 0; n0 < 8192; ++n0) s += __expf(p[(long long)n0 * 1024] - mx);
  cmax[cg] = mx;
  crsum[cg] = 1.f / s;
}

// ---------------------------------------------------------------------------
// Tiled transpose + convert: in fp32 [4][8192][1024] -> out f16 [4096][8192]
// (out row = b*1024 + channel). Optionally applies exp(x-max)*rsum (k path).
// ---------------------------------------------------------------------------
__global__ __launch_bounds__(256) void transpose_cvt(
    const float* __restrict__ in, f16* __restrict__ out,
    const float* __restrict__ cmax, const float* __restrict__ crsum) {
  __shared__ float tile[64][65];
  int b = blockIdx.z;
  int c0 = blockIdx.x * 64, n0 = blockIdx.y * 64;
  int tx = threadIdx.x & 63, ty = threadIdx.x >> 6;
  const float* src = in + (long long)b * 8192 * 1024;
#pragma unroll
  for (int i = ty; i < 64; i += 4)
    tile[i][tx] = src[(long long)(n0 + i) * 1024 + c0 + tx];
  __syncthreads();
#pragma unroll
  for (int i = ty; i < 64; i += 4) {
    int cg = b * 1024 + c0 + i;
    float x = tile[tx][i];
    if (cmax) x = __expf(x - cmax[cg]) * crsum[cg];
    out[(long long)cg * 8192 + n0 + tx] = (f16)x;
  }
}

// ---------------------------------------------------------------------------
// ctx transpose+convert: fp32 [32][128(dk)][128(dv)] -> f16 [32][dv][dk]
// ---------------------------------------------------------------------------
__global__ void ctx_transpose(const float* __restrict__ ctx,
                              f16* __restrict__ ctxT) {
  int z = blockIdx.x;
  const float* src = ctx + (long long)z * 16384;
  f16* dst = ctxT + (long long)z * 16384;
  for (int idx = threadIdx.x; idx < 16384; idx += blockDim.x) {
    int dv = idx & 127, dk = idx >> 7;
    dst[dv * 128 + dk] = (f16)src[idx];
  }
}

// ---------------------------------------------------------------------------
// Host-side orchestration
// ---------------------------------------------------------------------------
extern "C" void kernel_launch(void* const* d_in, const int* in_sizes, int n_in,
                              void* d_out, int out_size, void* d_ws,
                              size_t ws_size, hipStream_t stream) {
  (void)in_sizes; (void)n_in; (void)out_size; (void)ws_size;
  const float* x  = (const float*)d_in[0];
  const float* Wq = (const float*)d_in[1];
  const float* Wk = (const float*)d_in[2];
  const float* Wv = (const float*)d_in[3];
  const float* Wo = (const float*)d_in[4];
  const float* bo = (const float*)d_in[5];

  const long long BN = 32768;  // B*N rows
  const long long C  = 1024;
  char* ws = (char*)d_ws;
  size_t off = 0;
  auto take = [&](size_t bytes) -> char* {
    char* p = ws + off;
    off += (bytes + 255) & ~(size_t)255;
    return p;
  };
  float* P     = (float*)take((size_t)BN * C * 4);  // fp32 proj scratch (q->k->v)
  f16*   xb    = (f16*)take((size_t)BN * C * 2);    // x f16; reused as attended
  f16*   wqb   = (f16*)take((size_t)C * C * 2);
  f16*   wkb   = (f16*)take((size_t)C * C * 2);
  f16*   wvb   = (f16*)take((size_t)C * C * 2);
  f16*   wob   = (f16*)take((size_t)C * C * 2);
  f16*   qsoft = (f16*)take((size_t)BN * C * 2);
  f16*   expkT = (f16*)take((size_t)4096 * 8192 * 2);
  f16*   vT    = (f16*)take((size_t)4096 * 8192 * 2);
  float* cmax  = (float*)take(4096 * 4);
  float* crsum = (float*)take(4096 * 4);
  float* ctx   = (float*)take((size_t)32 * 128 * 128 * 4);
  f16*   ctxT  = (f16*)take((size_t)32 * 128 * 128 * 2);

  dim3 blk(256);

  // 1. convert inputs to f16
  cvt_f32_f16<<<4096, blk, 0, stream>>>(x, xb, BN * C);
  cvt_f32_f16<<<1024, blk, 0, stream>>>(Wq, wqb, C * C);
  cvt_f32_f16<<<1024, blk, 0, stream>>>(Wk, wkb, C * C);
  cvt_f32_f16<<<1024, blk, 0, stream>>>(Wv, wvb, C * C);
  cvt_f32_f16<<<1024, blk, 0, stream>>>(Wo, wob, C * C);

  // 2. q = x Wq^T (fp32) -> row softmax over each head's 128 channels (f16)
  gemm_tn<false, false><<<dim3(8, 256, 1), blk, 0, stream>>>(
      xb, 1024, 0, 0, wqb, 1024, 0, 0, P, 1024, 0, 0, nullptr,
      32768, 1024, 1024, 1);
  row_softmax_128<<<32768, blk, 0, stream>>>(P, qsoft, 262144);

  // 3. k = x Wk^T (fp32) -> column softmax over tokens, transposed f16
  gemm_tn<false, false><<<dim3(8, 256, 1), blk, 0, stream>>>(
      xb, 1024, 0, 0, wkb, 1024, 0, 0, P, 1024, 0, 0, nullptr,
      32768, 1024, 1024, 1);
  col_stats<<<16, blk, 0, stream>>>(P, cmax, crsum);
  transpose_cvt<<<dim3(16, 128, 4), blk, 0, stream>>>(P, expkT, cmax, crsum);

  // 4. v = x Wv^T (fp32) -> transposed f16
  gemm_tn<false, false><<<dim3(8, 256, 1), blk, 0, stream>>>(
      xb, 1024, 0, 0, wvb, 1024, 0, 0, P, 1024, 0, 0, nullptr,
      32768, 1024, 1024, 1);
  transpose_cvt<<<dim3(16, 128, 4), blk, 0, stream>>>(P, vT, nullptr, nullptr);

  // 5. ctx[z] = expkT[z] * vT[z]^T  (128x128, K=8192, z = b*8+h)
  gemm_tn<false, false><<<dim3(1, 1, 32), blk, 0, stream>>>(
      expkT, 8192, 1048576, 0, vT, 8192, 1048576, 0, ctx, 128, 16384, 0,
      nullptr, 128, 128, 8192, 1);
  ctx_transpose<<<32, blk, 0, stream>>>(ctx, ctxT);

  // 6. attended[b,:,h,:] = qsoft[b,:,h,:] * ctx[z]   (8192x128, K=128) -> xb
  gemm_tn<true, false><<<dim3(1, 64, 32), blk, 0, stream>>>(
      qsoft, 1024, 8388608LL, 128, ctxT, 128, 131072LL, 16384, xb, 1024,
      8388608LL, 128, nullptr, 8192, 128, 128, 8);

  // 7. out = attended Wo^T + bo (fp32 to d_out)
  gemm_tn<false, true><<<dim3(8, 256, 1), blk, 0, stream>>>(
      xb, 1024, 0, 0, wob, 1024, 0, 0, d_out, 1024, 0, 0, bo,
      32768, 1024, 1024, 1);
}
